// ToolAwareAttention_55465207661149
// MI455X (gfx1250) — compile-verified, matrix operand order checked by reading, not run on hardware
//
#include <hip/hip_runtime.h>

// ---------------------------------------------------------------------------
// ToolAwareAttention for MI455X (gfx1250, wave32, WMMA, async-to-LDS).
//
// y = sort_by_tool(x);  y1 = y @ W[tool] + b[tool];  out = y1 @ outW + outB
// Two 32768x1024x1024 GEMMs (~137 GFLOP) vs ~300MB traffic -> compute bound.
// bf16x3 split (hi+lo) on V_WMMA_F32_16X16X32_BF16: ~fp32 accuracy at ~2.7x
// the native F32-WMMA FLOP ceiling.  B tiles (and pass-2 A tiles, which are
// pre-split bf16) are staged with GLOBAL_LOAD_ASYNC_TO_LDS_B128, double
// buffered on ASYNCcnt.
// ---------------------------------------------------------------------------

#define HID   1024
#define NNODE 32768
#define TOOLS 8

#define BM 128          // block tile rows
#define BN 128          // block tile cols
#define BK 32           // k-step (one bf16 WMMA K)
#define LDSROW 40       // bf16 per LDS row: 32 data + 8 pad (80B stride, conflict-free)
#define HLSTRIDE (BM * LDSROW * 2)   // bytes between hi and lo planes in LDS

typedef __bf16 bf16;
typedef __attribute__((ext_vector_type(8)))  bf16  v8bf;
typedef __attribute__((ext_vector_type(16))) bf16  v16bf;
typedef __attribute__((ext_vector_type(8)))  float v8f;

#define S_WAIT_ASYNCCNT(N) asm volatile("s_wait_asynccnt " #N ::: "memory")

// One 16B chunk per lane, global -> LDS, tracked on ASYNCcnt (no VGPR data).
__device__ __forceinline__ void async_b128(unsigned lds, const bf16* base, unsigned voff) {
  asm volatile("global_load_async_to_lds_b128 %0, %1, %2"
               :: "v"(lds), "v"(voff), "s"(base) : "memory");
}

__device__ __forceinline__ unsigned lds_addr(const bf16* p) {
  return (unsigned)(size_t)p;   // low 32 bits of a flat LDS pointer = LDS byte address
}

// Stage a 128x32 bf16 tile pair (hi+lo planes) async into LDS.
// R0 = first global row (A rows or B cols); layout [row][k], k contiguous.
// 4 async b128 per thread (2 chunks x {hi,lo}).
__device__ __forceinline__ void stage_async_pair(unsigned ldsH, const bf16* gh,
                                                 const bf16* gl, int R0, int k0,
                                                 int tid) {
#pragma unroll
  for (int i = 0; i < 2; ++i) {
    int lin = tid + i * 256;          // 0..511 16B chunks
    int r   = lin >> 2;
    int seg = lin & 3;
    unsigned voff = (unsigned)(R0 + r) * (HID * 2u) + (unsigned)(k0 + seg * 8) * 2u;
    unsigned la   = (unsigned)(r * LDSROW + seg * 8) * 2u;
    async_b128(ldsH + la,            gh, voff);
    async_b128(ldsH + la + HLSTRIDE, gl, voff);
  }
}

// ---------------------------------------------------------------------------
// WMMA fragment load from LDS (CDNA5 ISA sec 7.12.2, 16-bit A/B 16x32):
//   lanes 0-15  : row r=lane,    K = 0..7 and 16..23
//   lanes 16-31 : row r=lane-16, K = 8..15 and 24..31
// ---------------------------------------------------------------------------
__device__ __forceinline__ v16bf load_frag(const bf16* rowPtr) {
  const int lane = threadIdx.x & 31;
  const int half = lane >> 4;
  const int r    = lane & 15;
  const bf16* p  = rowPtr + r * LDSROW + half * 8;
  v8bf a = *(const v8bf*)p;
  v8bf b = *(const v8bf*)(p + 16);
  return __builtin_shufflevector(a, b, 0,1,2,3,4,5,6,7,8,9,10,11,12,13,14,15);
}

__device__ __forceinline__ v8f wmma_bf16(v16bf a, v16bf b, v8f c) {
  return __builtin_amdgcn_wmma_f32_16x16x32_bf16(false, a, false, b,
                                                 (short)0, c, false, false);
}

// ---------------------------------------------------------------------------
// Kernel 1: stable counting sort by tool id + segment-aligned tile list.
// ---------------------------------------------------------------------------
__global__ __launch_bounds__(256)
void ta_build_perm(const int* __restrict__ toolIdx, int* __restrict__ perm,
                   int4* __restrict__ tiles, int* __restrict__ nTiles) {
  __shared__ int cnt[256][TOOLS];
  __shared__ int base[TOOLS];
  __shared__ int tot[TOOLS];
  const int tid = threadIdx.x;
  const int CH  = NNODE / 256;
  const int s   = tid * CH;

  int loc[TOOLS];
#pragma unroll
  for (int t = 0; t < TOOLS; ++t) loc[t] = 0;
  for (int i = 0; i < CH; ++i) loc[toolIdx[s + i] & (TOOLS - 1)]++;
#pragma unroll
  for (int t = 0; t < TOOLS; ++t) cnt[tid][t] = loc[t];
  __syncthreads();

  if (tid < TOOLS) {
    int run = 0;
    for (int c = 0; c < 256; ++c) { int v = cnt[c][tid]; cnt[c][tid] = run; run += v; }
    tot[tid] = run;
  }
  __syncthreads();

  if (tid == 0) {
    int run = 0, nt = 0;
    for (int t = 0; t < TOOLS; ++t) {
      base[t] = run;
      int c = tot[t], off = 0;
      while (off < c) {
        int rows = c - off; if (rows > BM) rows = BM;
        tiles[nt++] = make_int4(t, run + off, rows, 0);
        off += BM;
      }
      run += c;
    }
    *nTiles = nt;
  }
  __syncthreads();

#pragma unroll
  for (int t = 0; t < TOOLS; ++t) loc[t] = base[t] + cnt[tid][t];
  for (int i = 0; i < CH; ++i) {
    int t = toolIdx[s + i] & (TOOLS - 1);
    perm[loc[t]++] = s + i;
  }
}

// ---------------------------------------------------------------------------
// Kernel 2: transpose weights to [n][k] and split fp32 -> bf16 hi/lo.
// z = 0..7: tool_W[z]; z = 8: out_W.
// ---------------------------------------------------------------------------
__global__ __launch_bounds__(256)
void ta_pack_weights(const float* __restrict__ toolW, const float* __restrict__ outW,
                     bf16* __restrict__ Bh, bf16* __restrict__ Bl) {
  __shared__ float tile[32][33];
  const int z  = blockIdx.z;
  const float* src = (z < TOOLS) ? (toolW + (size_t)z * HID * HID) : outW;
  const size_t dstBase = (size_t)z * HID * HID;
  const int tx = threadIdx.x, ty = threadIdx.y;           // 32 x 8
  const int x0 = blockIdx.x * 32;                         // n
  const int y0 = blockIdx.y * 32;                         // kc
#pragma unroll
  for (int j = 0; j < 4; ++j)
    tile[ty + 8 * j][tx] = src[(size_t)(y0 + ty + 8 * j) * HID + x0 + tx];
  __syncthreads();
#pragma unroll
  for (int j = 0; j < 4; ++j) {
    float  w = tile[tx][ty + 8 * j];
    bf16   h = (bf16)w;
    size_t o = dstBase + (size_t)(x0 + ty + 8 * j) * HID + y0 + tx;
    Bh[o] = h;
    Bl[o] = (bf16)(w - (float)h);
  }
}

// ---------------------------------------------------------------------------
// Kernel 3: tiled GEMM, bf16x3 on v_wmma_f32_16x16x32_bf16.
// 256 threads = 8 waves; tile 128x128; wave -> 64x32 (4x2 frags).
//   pass 1 (Af != null): A = fp32 rows gathered via perm/tiles, split on the
//     fly (sync path); B async double-buffered; output split bf16 hi/lo.
//   pass 2 (Af == null): A = pre-split bf16 (Y1), A+B async double-buffered;
//     output fp32.
// ---------------------------------------------------------------------------
__global__ __launch_bounds__(256)
void ta_gemm(const float* __restrict__ Af,
             const bf16* __restrict__ Ah, const bf16* __restrict__ Al,
             const bf16* __restrict__ Bh, const bf16* __restrict__ Bl,
             const float* __restrict__ bias,
             float* __restrict__ OutF,
             bf16* __restrict__ OutH, bf16* __restrict__ OutL,
             const int*  __restrict__ perm,
             const int4* __restrict__ tiles,
             const int*  __restrict__ nTilesPtr) {
  __shared__ __align__(16) bf16 sA[2][2][BM * LDSROW];   // [buf][hi/lo][...]
  __shared__ __align__(16) bf16 sB[2][2][BN * LDSROW];
  __shared__ int sRow[BM];

  int tool, rowBase, nRows;
  if (tiles) {
    if ((int)blockIdx.x >= *nTilesPtr) return;
    int4 d = tiles[blockIdx.x];
    tool = d.x; rowBase = d.y; nRows = d.z;
  } else {
    tool = 0; rowBase = blockIdx.x * BM; nRows = BM;
  }

  const int tid     = threadIdx.x;
  const int colBase = blockIdx.y * BN;
  const bf16*  Bhm  = Bh + (size_t)tool * HID * HID;
  const bf16*  Blm  = Bl + (size_t)tool * HID * HID;
  const float* brow = bias + (size_t)tool * HID * (tiles ? 1 : 0);

  // Row map: out-of-range rows alias row 0 (their accumulators are never
  // stored; WMMA rows are independent) -> no exec-mask branch while staging.
  if (tid < BM)
    sRow[tid] = (tid < nRows) ? (perm ? perm[rowBase + tid] : (rowBase + tid)) : 0;
  __syncthreads();

  const unsigned aA[2] = { lds_addr(&sA[0][0][0]), lds_addr(&sA[1][0][0]) };
  const unsigned aB[2] = { lds_addr(&sB[0][0][0]), lds_addr(&sB[1][0][0]) };

  const int wave = tid >> 5;
  const int wm   = wave >> 2;        // 0..1 -> rows wm*64
  const int wn   = wave & 3;         // 0..3 -> cols wn*32

  v8f acc[4][2] = {};

  // --- prologue: async-issue buffer 0 -----------------------------------
  if (!Af) stage_async_pair(aA[0], Ah, Al, rowBase, 0, tid);
  stage_async_pair(aB[0], Bhm, Blm, colBase, 0, tid);

  for (int k0 = 0; k0 < HID; k0 += BK) {
    const int  cur     = (k0 >> 5) & 1;
    const bool hasNext = (k0 + BK) < HID;

    // async-issue next buffer while this one drains / A stages
    if (hasNext) {
      if (!Af) stage_async_pair(aA[cur ^ 1], Ah, Al, rowBase, k0 + BK, tid);
      stage_async_pair(aB[cur ^ 1], Bhm, Blm, colBase, k0 + BK, tid);
    }

    if (Af) {
      // pass 1: sync A stage with fp32 -> bf16 hi/lo split (overlaps async)
#pragma unroll
      for (int i = 0; i < 4; ++i) {
        int lin = tid + i * 256;       // 0..1023 float4 slots
        int r   = lin >> 3;
        int seg = lin & 7;
        float4 v = *(const float4*)(Af + (size_t)sRow[r] * HID + k0 + seg * 4);
        bf16* ph = &sA[cur][0][r * LDSROW + seg * 4];
        bf16* pl = &sA[cur][1][r * LDSROW + seg * 4];
        float xs[4] = {v.x, v.y, v.z, v.w};
#pragma unroll
        for (int j = 0; j < 4; ++j) {
          bf16 h = (bf16)xs[j];
          ph[j]  = h;
          pl[j]  = (bf16)(xs[j] - (float)h);
        }
      }
      // drain current B (4 in flight for next buffer while it has one)
      if (hasNext) S_WAIT_ASYNCCNT(4); else S_WAIT_ASYNCCNT(0);
    } else {
      // drain current A+B (8 in flight for next buffer)
      if (hasNext) S_WAIT_ASYNCCNT(8); else S_WAIT_ASYNCCNT(0);
    }
    __syncthreads();

    const bf16* pAh = &sA[cur][0][0];
    const bf16* pAl = &sA[cur][1][0];
    const bf16* pBh = &sB[cur][0][0];
    const bf16* pBl = &sB[cur][1][0];

    v16bf bh[2], bl[2];
#pragma unroll
    for (int ni = 0; ni < 2; ++ni) {
      bh[ni] = load_frag(pBh + (wn * 32 + ni * 16) * LDSROW);
      bl[ni] = load_frag(pBl + (wn * 32 + ni * 16) * LDSROW);
    }
#pragma unroll
    for (int mi = 0; mi < 4; ++mi) {
      v16bf ah = load_frag(pAh + (wm * 64 + mi * 16) * LDSROW);
      v16bf al = load_frag(pAl + (wm * 64 + mi * 16) * LDSROW);
#pragma unroll
      for (int ni = 0; ni < 2; ++ni) {
        acc[mi][ni] = wmma_bf16(al, bh[ni], acc[mi][ni]);  // lo*hi
        acc[mi][ni] = wmma_bf16(ah, bl[ni], acc[mi][ni]);  // hi*lo
        acc[mi][ni] = wmma_bf16(ah, bh[ni], acc[mi][ni]);  // hi*hi
      }
    }
    __syncthreads();
  }

  // --- store: C/D layout: VGPR j -> M = j + 8*(lane>=16), N = lane%16 -----
  const int lane = tid & 31;
  const int half = lane >> 4;
  const int c    = lane & 15;
#pragma unroll
  for (int mi = 0; mi < 4; ++mi)
#pragma unroll
    for (int ni = 0; ni < 2; ++ni) {
      int col  = colBase + wn * 32 + ni * 16 + c;
      float bv = brow[col];
#pragma unroll
      for (int j = 0; j < 8; ++j) {
        int r = wm * 64 + mi * 16 + half * 8 + j;
        if (r < nRows) {
          size_t o = (size_t)(rowBase + r) * HID + col;
          float  v = acc[mi][ni][j] + bv;
          if (OutF) {
            OutF[o] = v;
          } else {
            bf16 h = (bf16)v;
            OutH[o] = h;
            OutL[o] = (bf16)(v - (float)h);
          }
        }
      }
    }
}

// ---------------------------------------------------------------------------
// Workspace layout (bytes):
//   perm  int[N]           @ 0        (128 KB)
//   nTiles int             @ O_META
//   tiles int4[264]        @ O_TILES
//   Bhi   bf16[9*H*H]      @ O_BH     (18 MB)
//   Blo   bf16[9*H*H]      @ O_BL     (18 MB)
//   Y1hi  bf16[N*H]        @ O_Y1H    (64 MB)
//   Y1lo  bf16[N*H]        @ O_Y1L    (64 MB)     total ~164 MB
// ---------------------------------------------------------------------------
#define O_META  (131072)
#define O_TILES (131072 + 256)
#define O_BH    (135680)
#define O_BL    (O_BH  + 9ull * HID * HID * 2)
#define O_Y1H   (O_BL  + 9ull * HID * HID * 2)
#define O_Y1L   (O_Y1H + (unsigned long long)NNODE * HID * 2)

extern "C" void kernel_launch(void* const* d_in, const int* in_sizes, int n_in,
                              void* d_out, int out_size, void* d_ws, size_t ws_size,
                              hipStream_t stream) {
  (void)in_sizes; (void)n_in; (void)out_size; (void)ws_size;
  const float* node    = (const float*)d_in[0];
  const int*   toolIdx = (const int*)  d_in[1];
  const float* toolW   = (const float*)d_in[2];
  const float* toolB   = (const float*)d_in[3];
  const float* outW    = (const float*)d_in[4];
  const float* outB    = (const float*)d_in[5];
  float*       out     = (float*)d_out;
  char*        ws      = (char*)d_ws;

  int*  perm  = (int*) (ws);
  int*  nT    = (int*) (ws + O_META);
  int4* tiles = (int4*)(ws + O_TILES);
  bf16* Bh    = (bf16*)(ws + O_BH);
  bf16* Bl    = (bf16*)(ws + O_BL);
  bf16* Y1h   = (bf16*)(ws + O_Y1H);
  bf16* Y1l   = (bf16*)(ws + O_Y1L);

  ta_build_perm<<<1, 256, 0, stream>>>(toolIdx, perm, tiles, nT);
  ta_pack_weights<<<dim3(HID / 32, HID / 32, TOOLS + 1), dim3(32, 8), 0, stream>>>(
      toolW, outW, Bh, Bl);
  // Pass 1: MoE GEMM, fp32 A gathered+split on the fly, split-bf16 output.
  ta_gemm<<<dim3(NNODE / BM + TOOLS, HID / BN), 256, 0, stream>>>(
      node, nullptr, nullptr, Bh, Bl, toolB, nullptr, Y1h, Y1l, perm, tiles, nT);
  // Pass 2: output projection, fully async A+B staging, fp32 output.
  ta_gemm<<<dim3(NNODE / BM, HID / BN), 256, 0, stream>>>(
      nullptr, Y1h, Y1l, Bh + (size_t)TOOLS * HID * HID, Bl + (size_t)TOOLS * HID * HID,
      outB, out, nullptr, nullptr, nullptr, nullptr, nullptr);
}
